// Decoder_45810121179438
// MI455X (gfx1250) — matrix operand order repaired and checked by reading, not compiled, not run on hardware
//
#include <hip/hip_runtime.h>
#include <hip/hip_bf16.h>

// ---------- types ----------
typedef __attribute__((ext_vector_type(16))) __bf16        v16bf;
typedef __attribute__((ext_vector_type(8)))  float         v8f;
typedef __attribute__((ext_vector_type(4)))  unsigned int  u32x4;

#define TLEN 24
#define BATCH 64
#define SRC 48
#define DMODEL 1024
#define EMBD 512

#define WMMA_BF16(a, b, c) \
    __builtin_amdgcn_wmma_f32_16x16x32_bf16(false, (a), false, (b), (short)0, (c), false, false)

__device__ __forceinline__ unsigned short f2bf(float f) {
    unsigned int u = __builtin_bit_cast(unsigned int, f);
    unsigned int r = u + 0x7fffu + ((u >> 16) & 1u);   // round-to-nearest-even
    return (unsigned short)(r >> 16);
}
__device__ __forceinline__ float sigmoidf(float x) { return 1.0f / (1.0f + __expf(-x)); }

// A fragment: 16x32 bf16, lane holds row m = mt*16 + (lane&15)
//   VGPR0-3: k = kb + (lane>>4)*16 + 0..7 ; VGPR4-7: k = kb + (lane>>4)*16 + 8..15
__device__ __forceinline__ v16bf ldA(const unsigned short* __restrict__ A, int K,
                                     int mt, int kb, int lane) {
    int m  = mt * 16 + (lane & 15);
    int hk = (lane >> 4) << 4;                 // 0 or 16
    const unsigned short* p = A + (size_t)m * K + kb + hk;
    u32x4 lo = *(const u32x4*)p;               // k .. k+7
    u32x4 hi = *(const u32x4*)(p + 8);         // k+8 .. k+15
    return __builtin_bit_cast(v16bf, __builtin_shufflevector(lo, hi, 0,1,2,3,4,5,6,7));
}
// B fragment: 32x16 bf16, lane holds col n = (lane&15); k = kb + (lane>>4)*16 + 0..15
// wrow = W + n*K + ((lane>>4)<<4)  (precomputed); 32 contiguous bytes at +kb
__device__ __forceinline__ v16bf ldB(const unsigned short* __restrict__ wrow, int kb) {
    const unsigned short* p = wrow + kb;
    u32x4 lo = *(const u32x4*)p;               // k .. k+7
    u32x4 hi = *(const u32x4*)(p + 8);         // k+8 .. k+15
    return __builtin_bit_cast(v16bf, __builtin_shufflevector(lo, hi, 0,1,2,3,4,5,6,7));
}

// ---------- weight convert: dst[n][0:KA+KB] = bf16([A[n]|B[n]]) ----------
__global__ void k_cvt_cat(const float* __restrict__ A, const float* __restrict__ B,
                          unsigned short* __restrict__ dst, int KA, int KB, int N) {
    int Kt = KA + KB;
    int total = N * Kt;
    for (int idx = blockIdx.x * 256 + threadIdx.x; idx < total; idx += gridDim.x * 256) {
        int row = idx / Kt, col = idx - row * Kt;
        float v = (col < KA) ? A[(size_t)row * KA + col]
                             : B[(size_t)row * KB + (col - KA)];
        dst[idx] = f2bf(v);
    }
}

// ---------- per-step input assembly ----------
__global__ void k_prep_x0(const int* __restrict__ tok, const float* __restrict__ emb,
                          const float* __restrict__ out_prev, const float* __restrict__ h_prev,
                          unsigned short* __restrict__ x) {
    int idx = blockIdx.x * 256 + threadIdx.x;              // 64*2560
    int b = idx / 2560, c = idx - b * 2560;
    float v;
    if (c < EMBD)             v = emb[(size_t)tok[b] * EMBD + c];
    else if (c < EMBD+DMODEL) v = out_prev[b * DMODEL + (c - EMBD)];
    else                      v = h_prev[b * DMODEL + (c - EMBD - DMODEL)];
    x[idx] = f2bf(v);
}
__global__ void k_prep_x1(const float* __restrict__ h0n, const float* __restrict__ h1p,
                          unsigned short* __restrict__ x) {
    int idx = blockIdx.x * 256 + threadIdx.x;              // 64*2048
    int b = idx >> 11, c = idx & 2047;
    float v = (c < DMODEL) ? h0n[b * DMODEL + c] : h1p[b * DMODEL + (c - DMODEL)];
    x[idx] = f2bf(v);
}
__global__ void k_cvt_h1(const float* __restrict__ h1, unsigned short* __restrict__ h1bf,
                         unsigned short* __restrict__ xout) {
    int idx = blockIdx.x * 256 + threadIdx.x;              // 64*1024
    int b = idx >> 10, j = idx & 1023;
    unsigned short u = f2bf(h1[idx]);
    h1bf[idx] = u;
    xout[b * 2048 + DMODEL + j] = u;
}

// ---------- fused GEMM + LSTM cell ----------
// A:[64,K] bf16, W:[4096,K] bf16 (rows = gate*1024 + n). Block = 16 N-cols, 4 waves (one/gate).
// K-loop is software-pipelined: next iteration's 10 b128 loads are issued before the
// 4 WMMAs consuming the current fragments, so waits are partial and latency overlaps XDL.
__global__ void k_gemm_lstm(const unsigned short* __restrict__ A,
                            const unsigned short* __restrict__ W, int K,
                            const float* __restrict__ b_ih, const float* __restrict__ b_hh,
                            const float* __restrict__ c_prev,
                            float* __restrict__ c_out, float* __restrict__ h_out) {
    __shared__ float gbuf[4][64][16];
    int tid  = threadIdx.x;
    int lane = tid & 31;
    int wave = tid >> 5;                                   // gate id
    int ncol0 = blockIdx.x * 16;
    int n = (wave << 10) + ncol0 + (lane & 15);
    const unsigned short* wrow = W + (size_t)n * K + ((lane >> 4) << 4);

    v8f acc0 = {0,0,0,0,0,0,0,0}, acc1 = acc0, acc2 = acc0, acc3 = acc0;

    v16bf b  = ldB(wrow, 0);
    v16bf a0 = ldA(A, K, 0, 0, lane);
    v16bf a1 = ldA(A, K, 1, 0, lane);
    v16bf a2 = ldA(A, K, 2, 0, lane);
    v16bf a3 = ldA(A, K, 3, 0, lane);
    for (int kb = 32; kb < K; kb += 32) {
        v16bf bn  = ldB(wrow, kb);
        v16bf an0 = ldA(A, K, 0, kb, lane);
        v16bf an1 = ldA(A, K, 1, kb, lane);
        v16bf an2 = ldA(A, K, 2, kb, lane);
        v16bf an3 = ldA(A, K, 3, kb, lane);
        acc0 = WMMA_BF16(a0, b, acc0);
        acc1 = WMMA_BF16(a1, b, acc1);
        acc2 = WMMA_BF16(a2, b, acc2);
        acc3 = WMMA_BF16(a3, b, acc3);
        b = bn; a0 = an0; a1 = an1; a2 = an2; a3 = an3;
    }
    acc0 = WMMA_BF16(a0, b, acc0);
    acc1 = WMMA_BF16(a1, b, acc1);
    acc2 = WMMA_BF16(a2, b, acc2);
    acc3 = WMMA_BF16(a3, b, acc3);

    int hi = lane >> 4, lo16 = lane & 15;
    #pragma unroll
    for (int r = 0; r < 8; ++r) {
        gbuf[wave][ 0 + 8*hi + r][lo16] = acc0[r];
        gbuf[wave][16 + 8*hi + r][lo16] = acc1[r];
        gbuf[wave][32 + 8*hi + r][lo16] = acc2[r];
        gbuf[wave][48 + 8*hi + r][lo16] = acc3[r];
    }
    __syncthreads();
    #pragma unroll
    for (int e = 0; e < 8; ++e) {
        int p = tid + 128 * e;                             // 0..1023
        int m = p >> 4, j = p & 15;
        int col = ncol0 + j;
        float gi = gbuf[0][m][j] + b_ih[col]          + b_hh[col];
        float gf = gbuf[1][m][j] + b_ih[1024 + col]   + b_hh[1024 + col];
        float gg = gbuf[2][m][j] + b_ih[2048 + col]   + b_hh[2048 + col];
        float go = gbuf[3][m][j] + b_ih[3072 + col]   + b_hh[3072 + col];
        int o = m * DMODEL + col;
        float c = sigmoidf(gf) * c_prev[o] + sigmoidf(gi) * tanhf(gg);
        c_out[o] = c;
        h_out[o] = sigmoidf(go) * tanhf(c);
    }
}

// ---------- plain GEMM: out[64,1024] = A[64,K] @ W[1024,K]^T (+tanh) ----------
__global__ void k_gemm(const unsigned short* __restrict__ A,
                       const unsigned short* __restrict__ W, int K,
                       float* __restrict__ out, int do_tanh) {
    int tid  = threadIdx.x;
    int lane = tid & 31;
    int wave = tid >> 5;
    int ntile = blockIdx.x * 64 + wave * 16;
    int n = ntile + (lane & 15);
    const unsigned short* wrow = W + (size_t)n * K + ((lane >> 4) << 4);

    v8f acc0 = {0,0,0,0,0,0,0,0}, acc1 = acc0, acc2 = acc0, acc3 = acc0;

    v16bf b  = ldB(wrow, 0);
    v16bf a0 = ldA(A, K, 0, 0, lane);
    v16bf a1 = ldA(A, K, 1, 0, lane);
    v16bf a2 = ldA(A, K, 2, 0, lane);
    v16bf a3 = ldA(A, K, 3, 0, lane);
    for (int kb = 32; kb < K; kb += 32) {
        v16bf bn  = ldB(wrow, kb);
        v16bf an0 = ldA(A, K, 0, kb, lane);
        v16bf an1 = ldA(A, K, 1, kb, lane);
        v16bf an2 = ldA(A, K, 2, kb, lane);
        v16bf an3 = ldA(A, K, 3, kb, lane);
        acc0 = WMMA_BF16(a0, b, acc0);
        acc1 = WMMA_BF16(a1, b, acc1);
        acc2 = WMMA_BF16(a2, b, acc2);
        acc3 = WMMA_BF16(a3, b, acc3);
        b = bn; a0 = an0; a1 = an1; a2 = an2; a3 = an3;
    }
    acc0 = WMMA_BF16(a0, b, acc0);
    acc1 = WMMA_BF16(a1, b, acc1);
    acc2 = WMMA_BF16(a2, b, acc2);
    acc3 = WMMA_BF16(a3, b, acc3);

    int hi = lane >> 4, lo16 = lane & 15;
    int ncol = ntile + lo16;
    #pragma unroll
    for (int r = 0; r < 8; ++r) {
        float v0 = acc0[r], v1 = acc1[r], v2 = acc2[r], v3 = acc3[r];
        if (do_tanh) { v0 = tanhf(v0); v1 = tanhf(v1); v2 = tanhf(v2); v3 = tanhf(v3); }
        out[( 0 + 8*hi + r) * DMODEL + ncol] = v0;
        out[(16 + 8*hi + r) * DMODEL + ncol] = v1;
        out[(32 + 8*hi + r) * DMODEL + ncol] = v2;
        out[(48 + 8*hi + r) * DMODEL + ncol] = v3;
    }
}

// ---------- attention small ops ----------
__global__ void k_scores(const float* __restrict__ ctx, const float* __restrict__ q,
                         float* __restrict__ scores) {
    int tid = threadIdx.x, lane = tid & 31;
    int w = blockIdx.x * 4 + (tid >> 5);                   // 0..3071
    int b = w / SRC, s = w - b * SRC;
    const float* cp = ctx + ((size_t)s * BATCH + b) * DMODEL;
    const float* qp = q + (size_t)b * DMODEL;
    float sum = 0.0f;
    for (int d = lane; d < DMODEL; d += 32) sum += cp[d] * qp[d];
    #pragma unroll
    for (int off = 16; off >= 1; off >>= 1) sum += __shfl_down(sum, off, 32);
    if (lane == 0) scores[b * SRC + s] = sum;
}

__global__ void k_softmax(const float* __restrict__ scores,
                          float* __restrict__ attn, float* __restrict__ attn_out) {
    __shared__ float buf[64];
    int b = blockIdx.x, s = threadIdx.x;
    float v = (s < SRC) ? scores[b * SRC + s] : -3.0e38f;
    buf[s] = v; __syncthreads();
    for (int o = 32; o >= 1; o >>= 1) { if (s < o) buf[s] = fmaxf(buf[s], buf[s + o]); __syncthreads(); }
    float mx = buf[0]; __syncthreads();
    float e = (s < SRC) ? __expf(v - mx) : 0.0f;
    buf[s] = e; __syncthreads();
    for (int o = 32; o >= 1; o >>= 1) { if (s < o) buf[s] += buf[s + o]; __syncthreads(); }
    float p = e / buf[0];
    if (s < SRC) { attn[b * SRC + s] = p; attn_out[b * SRC + s] = p; }
}

__global__ void k_wctx(const float* __restrict__ attn, const float* __restrict__ ctx,
                       unsigned short* __restrict__ xout) {
    int idx = blockIdx.x * 256 + threadIdx.x;              // 64*1024
    int b = idx >> 10, d = idx & 1023;
    const float* ap = attn + b * SRC;
    float sum = 0.0f;
    for (int s = 0; s < SRC; ++s) sum += ap[s] * ctx[((size_t)s * BATCH + b) * DMODEL + d];
    xout[b * 2048 + d] = f2bf(sum);
}

// ---------- host ----------
extern "C" void kernel_launch(void* const* d_in, const int* in_sizes, int n_in,
                              void* d_out, int out_size, void* d_ws, size_t ws_size,
                              hipStream_t stream) {
    const int*   tokens  = (const int*)  d_in[0];
    const float* context = (const float*)d_in[1];
    const float* h0      = (const float*)d_in[2];
    const float* c0      = (const float*)d_in[3];
    const float* init_o  = (const float*)d_in[4];
    const float* emb     = (const float*)d_in[5];
    const float* W_ih_0  = (const float*)d_in[6];
    const float* W_hh_0  = (const float*)d_in[7];
    const float* b_ih_0  = (const float*)d_in[8];
    const float* b_hh_0  = (const float*)d_in[9];
    const float* W_ih_1  = (const float*)d_in[10];
    const float* W_hh_1  = (const float*)d_in[11];
    const float* b_ih_1  = (const float*)d_in[12];
    const float* b_hh_1  = (const float*)d_in[13];
    const float* W_in    = (const float*)d_in[14];
    const float* W_out   = (const float*)d_in[15];

    float* outputs  = (float*)d_out;                         // [24][64][1024]
    float* hT       = outputs + TLEN * BATCH * DMODEL;       // [2][64][1024]
    float* cT       = hT + 2 * BATCH * DMODEL;               // [2][64][1024]
    float* attn_out = cT + 2 * BATCH * DMODEL;               // [64][48]

    char* wp = (char*)d_ws;
    auto alloc = [&](size_t bytes) { char* p = wp; wp += (bytes + 255) & ~(size_t)255; return p; };
    unsigned short* Wcat0 = (unsigned short*)alloc((size_t)4096 * 2560 * 2);
    unsigned short* Wcat1 = (unsigned short*)alloc((size_t)4096 * 2048 * 2);
    unsigned short* Winb  = (unsigned short*)alloc((size_t)1024 * 1024 * 2);
    unsigned short* Woutb = (unsigned short*)alloc((size_t)1024 * 2048 * 2);
    unsigned short* xcat0 = (unsigned short*)alloc((size_t)BATCH * 2560 * 2);
    unsigned short* xcat1 = (unsigned short*)alloc((size_t)BATCH * 2048 * 2);
    unsigned short* h1bf  = (unsigned short*)alloc((size_t)BATCH * 1024 * 2);
    unsigned short* xoutb = (unsigned short*)alloc((size_t)BATCH * 2048 * 2);
    float* qbuf  = (float*)alloc((size_t)BATCH * 1024 * 4);
    float* sbuf  = (float*)alloc((size_t)BATCH * SRC * 4);
    float* abuf  = (float*)alloc((size_t)BATCH * SRC * 4);

    // one-time weight conversion (bf16, [W_ih|W_hh] fused per row)
    k_cvt_cat<<<8192, 256, 0, stream>>>(W_ih_0, W_hh_0, Wcat0, 1536, 1024, 4096);
    k_cvt_cat<<<8192, 256, 0, stream>>>(W_ih_1, W_hh_1, Wcat1, 1024, 1024, 4096);
    k_cvt_cat<<<4096, 256, 0, stream>>>(W_in,  W_in,  Winb,  1024, 0, 1024);
    k_cvt_cat<<<4096, 256, 0, stream>>>(W_out, W_out, Woutb, 2048, 0, 1024);

    const int BD = BATCH * DMODEL;                           // 65536
    for (int t = 0; t < TLEN; ++t) {
        const float* out_prev = t ? outputs + (size_t)(t - 1) * BD : init_o;
        const float* h0_prev  = t ? hT        : h0;
        const float* h1_prev  = t ? hT + BD   : h0 + BD;
        const float* c0_prev  = t ? cT        : c0;
        const float* c1_prev  = t ? cT + BD   : c0 + BD;

        k_prep_x0<<<640, 256, 0, stream>>>(tokens + t * BATCH, emb, out_prev, h0_prev, xcat0);
        k_gemm_lstm<<<64, 128, 0, stream>>>(xcat0, Wcat0, 2560, b_ih_0, b_hh_0, c0_prev, cT, hT);
        k_prep_x1<<<512, 256, 0, stream>>>(hT, h1_prev, xcat1);
        k_gemm_lstm<<<64, 128, 0, stream>>>(xcat1, Wcat1, 2048, b_ih_1, b_hh_1, c1_prev, cT + BD, hT + BD);
        k_cvt_h1<<<256, 256, 0, stream>>>(hT + BD, h1bf, xoutb);
        k_gemm<<<16, 128, 0, stream>>>(h1bf, Winb, 1024, qbuf, 0);
        k_scores<<<768, 128, 0, stream>>>(context, qbuf, sbuf);
        k_softmax<<<64, 64, 0, stream>>>(sbuf, abuf, attn_out);
        k_wctx<<<256, 256, 0, stream>>>(abuf, context, xoutb);
        k_gemm<<<16, 128, 0, stream>>>(xoutb, Woutb, 2048, outputs + (size_t)t * BD, 1);
    }
}
